// SMPE2_Encoder_1030792151096
// MI455X (gfx1250) — matrix operand order repaired and checked by reading, not compiled, not run on hardware
//
#include <hip/hip_runtime.h>

// ---------------------------------------------------------------------------
// MI455X (gfx1250) implementation of the SMPE2 encoder reference.
// Wave32; one wave handles 16 batch rows end-to-end. All GEMM-shaped work
// uses v_wmma_f32_16x16x32_bf16 (f32 accumulate); reductions/gates in f32.
// Memory-bound workload (~540MB traffic vs ~64 GFLOP) -> single streaming
// pass over obs/act; weights converted once to bf16 in d_ws.
// ---------------------------------------------------------------------------

typedef __bf16 bf16;
typedef __attribute__((ext_vector_type(16))) __bf16 v16bf;
typedef __attribute__((ext_vector_type(8)))  __bf16 v8bf;
typedef __attribute__((ext_vector_type(8)))  float  v8f;

#define DEVINL static __device__ __forceinline__

DEVINL v8f wmma_bf(v16bf a, v16bf b, v8f c) {
  return __builtin_amdgcn_wmma_f32_16x16x32_bf16(false, a, false, b, (short)0, c,
                                                 false, false);
}

DEVINL v8f splat8(float v) {
  v8f r;
#pragma unroll
  for (int j = 0; j < 8; ++j) r[j] = v;
  return r;
}

DEVINL float wsum(float v) {
#pragma unroll
  for (int o = 16; o > 0; o >>= 1) v += __shfl_xor(v, o, 32);
  return v;
}
DEVINL float wmaxr(float v) {
#pragma unroll
  for (int o = 16; o > 0; o >>= 1) v = fmaxf(v, __shfl_xor(v, o, 32));
  return v;
}
DEVINL float sigmf(float x) { return 1.0f / (1.0f + __expf(-x)); }

// B fragment (32x16 bf16) from row-major bf16 weight W[N][K]; computes A*W^T.
// Layout: lanes 0-15 hold K=0..7,16..23 (packed pairs over 8 VGPRs), lanes 16-31
// hold K=8..15,24..31; lane&15 = output column N.
DEVINL v16bf load_bfrag(const bf16* __restrict__ W, int K, int ntile, int ktile,
                        int lane) {
  const int n  = ntile * 16 + (lane & 15);
  const int kb = ktile * 32 + ((lane & 16) ? 8 : 0);
  const bf16* p = W + (size_t)n * K + kb;
  const v8bf lo = *(const v8bf*)(p);
  const v8bf hi = *(const v8bf*)(p + 16);
  v16bf r;
#pragma unroll
  for (int i = 0; i < 8; ++i) { r[i] = lo[i]; r[i + 8] = hi[i]; }
  return r;
}

// A fragment (16x32 bf16) from f32 row pointer (rowp points at this K-chunk's
// k=0 for row lane&15). Lanes<16: elems0-7=K0..7, elems8-15=K16..23;
// lanes>=16: K8..15 and K24..31.
DEVINL v16bf load_afrag_f32(const float* __restrict__ rowp, int lane) {
  const int kb = (lane & 16) ? 8 : 0;
  v16bf r;
#pragma unroll
  for (int i = 0; i < 8; ++i) {
    r[i]     = (bf16)rowp[kb + i];
    r[i + 8] = (bf16)rowp[kb + 16 + i];
  }
  return r;
}

// A fragment from bf16 LDS matrix [16][H], row-major.
DEVINL v16bf load_afrag_lds(const bf16* base, int H, int ktile, int lane) {
  const int row = lane & 15;
  const int kb  = ktile * 32 + ((lane & 16) ? 8 : 0);
  const bf16* p = base + row * H + kb;
  v16bf r;
#pragma unroll
  for (int i = 0; i < 8; ++i) { r[i] = p[i]; r[i + 8] = p[16 + i]; }
  return r;
}

// Store f32 D fragment (16x16) to bf16 LDS matrix [16][H] at column base.
// D layout: elem j -> row M=j+((lane&16)?8:0), col N=lane&15.
DEVINL void store_dfrag_bf16(bf16* base, int H, int colbase, v8f d, int lane) {
  const int col  = colbase + (lane & 15);
  const int roff = (lane & 16) ? 8 : 0;
#pragma unroll
  for (int j = 0; j < 8; ++j) base[(roff + j) * H + col] = (bf16)d[j];
}

struct SmpeParams {
  const float* obs; const float* act;
  const float* hist_bih; const float* hist_bhh;
  const float* ln_g; const float* ln_b;
  const float* dtw_W1; const float* dtw_b1;
  const float* dtw_W2; const float* dtw_b2;
  const float* dtw_W3; const float* dtw_b3;
  const float* gf_bih; const float* gf_bhh;
  const float* gb_bih; const float* gb_bhh;
  const float* bi_b1; const float* bi_b2;
  const float* mu_b; const float* lv_b;
  const bf16* w_histWih; const bf16* w_histWhh;
  const bf16* w_gfWih;   const bf16* w_gfWhh;
  const bf16* w_gbWih;   const bf16* w_gbWhh;
  const bf16* w_biW1;    const bf16* w_biW2;
  const bf16* w_muW;     const bf16* w_lvW;
  float* out;
  int B;
};

// Bidirectional-window GRU: HE=64, 8 masked steps, per-row time gather.
// forward: time = 14 - off[row] + t ; backward: time = 14 + off[row] - t.
// Times are always in [0,21]; masked (t>=len) steps are harmless garbage.
DEVINL void run_gru8(const SmpeParams& p, const bf16* Wih, const bf16* Whh,
                     const float* bih, const float* bhh, const int* s_off,
                     bool forward, bf16* s_h, int lane, int b0, v8f h[4]) {
#pragma unroll
  for (int tt = 0; tt < 4; ++tt) h[tt] = splat8(0.0f);
  const int roff = (lane & 16) ? 8 : 0;
  int lens[8];
#pragma unroll
  for (int j = 0; j < 8; ++j) lens[j] = s_off[roff + j] + 1;
  const int arow = lane & 15;
  const int aoff = s_off[arow];

  for (int t = 0; t < 8; ++t) {
    const int tm = forward ? (14 - aoff + t) : (14 + aoff - t);
    v8f X[12], Gn[4];
#pragma unroll
    for (int nt = 0; nt < 12; ++nt) {
      const int col = nt * 16 + (lane & 15);
      X[nt] = splat8(bih[col] + (nt < 8 ? bhh[col] : 0.0f));
    }
#pragma unroll
    for (int tt = 0; tt < 4; ++tt)
      Gn[tt] = splat8(bhh[128 + tt * 16 + (lane & 15)]);

    const float* orow = p.obs + ((size_t)(b0 + arow) * 22 + tm) * 128;
    const float* arw  = p.act + ((size_t)(b0 + arow) * 22 + tm) * 32;
#pragma unroll
    for (int kt = 0; kt < 5; ++kt) {
      const float* base = (kt < 4) ? (orow + kt * 32) : arw;
      const v16bf A = load_afrag_f32(base, lane);
#pragma unroll
      for (int nt = 0; nt < 12; ++nt)
        X[nt] = wmma_bf(A, load_bfrag(Wih, 160, nt, kt, lane), X[nt]);
    }

    __syncthreads();
#pragma unroll
    for (int tt = 0; tt < 4; ++tt) store_dfrag_bf16(s_h, 64, tt * 16, h[tt], lane);
    __syncthreads();
#pragma unroll
    for (int kt = 0; kt < 2; ++kt) {
      const v16bf Ah = load_afrag_lds(s_h, 64, kt, lane);
#pragma unroll
      for (int nt = 0; nt < 12; ++nt) {
        const v16bf Bw = load_bfrag(Whh, 64, nt, kt, lane);
        if (nt < 8) X[nt] = wmma_bf(Ah, Bw, X[nt]);
        else        Gn[nt - 8] = wmma_bf(Ah, Bw, Gn[nt - 8]);
      }
    }
#pragma unroll
    for (int tt = 0; tt < 4; ++tt) {
#pragma unroll
      for (int j = 0; j < 8; ++j) {
        const float rr = sigmf(X[tt][j]);
        const float zz = sigmf(X[tt + 4][j]);
        const float nn = tanhf(X[tt + 8][j] + rr * Gn[tt][j]);
        const float hn = (1.0f - zz) * nn + zz * h[tt][j];
        h[tt][j] = (t < lens[j]) ? hn : h[tt][j];
      }
    }
  }
}

__global__ __launch_bounds__(32) void smpe2_main(SmpeParams p) {
  const int lane = threadIdx.x;
  const int b0 = blockIdx.x * 16;

  __shared__ float s_ent[16], s_roc[16], s_corr[16];
  __shared__ float s_hfeat[16 * 32];
  __shared__ int   s_soff[16], s_eoff[16];
  __shared__ bf16  s_h[16 * 64];
  __shared__ bf16  s_comb[16 * 128];
  __shared__ bf16  s_tmp[16 * 64];

  // ---------------- Phase 1: entropy / roc / corr (per 16 rows) ------------
  for (int r = 0; r < 16; ++r) {
    const float* orow = p.obs + ((size_t)(b0 + r) * 22 + 14) * 128;
    const float x0 = orow[lane], x1 = orow[lane + 32];
    const float x2 = orow[lane + 64], x3 = orow[lane + 96];
    const float m = wmaxr(fmaxf(fmaxf(x0, x1), fmaxf(x2, x3)));
    const float e0 = __expf(x0 - m), e1 = __expf(x1 - m);
    const float e2 = __expf(x2 - m), e3 = __expf(x3 - m);
    const float s = wsum(e0 + e1 + e2 + e3);
    const float inv = 1.0f / s;
    const float p0 = e0 * inv, p1 = e1 * inv, p2 = e2 * inv, p3 = e3 * inv;
    float part = p0 * logf(p0 + 1e-8f) + p1 * logf(p1 + 1e-8f) +
                 p2 * logf(p2 + 1e-8f) + p3 * logf(p3 + 1e-8f);
    const float ent = -wsum(part);

    float acc = 0.0f;
    for (int d = 0; d < 3; ++d) {
      const float* ra = p.obs + ((size_t)(b0 + r) * 22 + 11 + d) * 128;
      const float* rb = ra + 128;
      float s2 = 0.0f;
#pragma unroll
      for (int j = 0; j < 4; ++j) {
        const float dd = rb[lane + 32 * j] - ra[lane + 32 * j];
        s2 += dd * dd;
      }
      acc += sqrtf(wsum(s2));
    }

    const float* arw = p.act + ((size_t)(b0 + r) * 22 + 13) * 32;
    const float a0 = arw[lane];  // padded act: nonzero only for k<32 (j==0)
    const float om = wsum(x0 + x1 + x2 + x3) * (1.0f / 128.0f);
    const float am = wsum(a0) * (1.0f / 128.0f);
    const float oc0 = x0 - om, oc1 = x1 - om, oc2 = x2 - om, oc3 = x3 - om;
    const float ac0 = a0 - am, acr = -am;
    const float dot = wsum(oc0 * ac0 + (oc1 + oc2 + oc3) * acr);
    const float no = sqrtf(wsum(oc0 * oc0 + oc1 * oc1 + oc2 * oc2 + oc3 * oc3));
    const float na = sqrtf(wsum(ac0 * ac0 + 3.0f * acr * acr));
    const float corr = dot / (no * na + 1e-8f);

    if (lane == 0) {
      s_ent[r] = ent; s_roc[r] = acc * (1.0f / 3.0f); s_corr[r] = corr;
    }
  }
  __syncthreads();

  // ---------------- Phase 2: history GRU (HG=32, 15 steps) -----------------
  {
    v8f h[2];
    h[0] = splat8(0.0f); h[1] = splat8(0.0f);
    for (int t = 0; t < 15; ++t) {
      v8f X[6], Gn[2];
#pragma unroll
      for (int nt = 0; nt < 6; ++nt) {
        const int col = nt * 16 + (lane & 15);
        X[nt] = splat8(p.hist_bih[col] + (nt < 4 ? p.hist_bhh[col] : 0.0f));
      }
#pragma unroll
      for (int tt = 0; tt < 2; ++tt)
        Gn[tt] = splat8(p.hist_bhh[64 + tt * 16 + (lane & 15)]);

      const int arow = lane & 15;
      const float* orow = p.obs + ((size_t)(b0 + arow) * 22 + t) * 128;
      const float* arw  = p.act + ((size_t)(b0 + arow) * 22 + t) * 32;
#pragma unroll
      for (int kt = 0; kt < 5; ++kt) {
        const float* base = (kt < 4) ? (orow + kt * 32) : arw;
        const v16bf A = load_afrag_f32(base, lane);
#pragma unroll
        for (int nt = 0; nt < 6; ++nt)
          X[nt] = wmma_bf(A, load_bfrag(p.w_histWih, 160, nt, kt, lane), X[nt]);
      }

      __syncthreads();
      store_dfrag_bf16(s_h, 32, 0, h[0], lane);
      store_dfrag_bf16(s_h, 32, 16, h[1], lane);
      __syncthreads();
      const v16bf Ah = load_afrag_lds(s_h, 32, 0, lane);
#pragma unroll
      for (int nt = 0; nt < 6; ++nt) {
        const v16bf Bw = load_bfrag(p.w_histWhh, 32, nt, 0, lane);
        if (nt < 4) X[nt] = wmma_bf(Ah, Bw, X[nt]);
        else        Gn[nt - 4] = wmma_bf(Ah, Bw, Gn[nt - 4]);
      }
#pragma unroll
      for (int tt = 0; tt < 2; ++tt) {
#pragma unroll
        for (int j = 0; j < 8; ++j) {
          const float rr = sigmf(X[tt][j]);
          const float zz = sigmf(X[tt + 2][j]);
          const float nn = tanhf(X[tt + 4][j] + rr * Gn[tt][j]);
          h[tt][j] = (1.0f - zz) * nn + zz * h[tt][j];
        }
      }
    }
    // hfeat -> LDS (f32, [16][32])
    const int roff = (lane & 16) ? 8 : 0;
#pragma unroll
    for (int tt = 0; tt < 2; ++tt) {
      const int col = tt * 16 + (lane & 15);
#pragma unroll
      for (int j = 0; j < 8; ++j) s_hfeat[(roff + j) * 32 + col] = h[tt][j];
    }
  }
  __syncthreads();

  // ---------------- Phase 3: layernorm + dtw MLP + argmax ------------------
  if (lane < 16) {
    const int r = lane;
    float f[35];
    f[0] = s_ent[r]; f[1] = s_roc[r]; f[2] = s_corr[r];
    for (int i = 0; i < 32; ++i) f[3 + i] = s_hfeat[r * 32 + i];
    float mean = 0.0f;
    for (int i = 0; i < 35; ++i) mean += f[i];
    mean *= (1.0f / 35.0f);
    float var = 0.0f;
    for (int i = 0; i < 35; ++i) { const float d = f[i] - mean; var += d * d; }
    var *= (1.0f / 35.0f);
    const float inv = 1.0f / sqrtf(var + 1e-5f);
    float fn[35];
    for (int i = 0; i < 35; ++i)
      fn[i] = (f[i] - mean) * inv * p.ln_g[i] + p.ln_b[i];
    float h1[64];
    for (int n = 0; n < 64; ++n) {
      float sacc = p.dtw_b1[n];
      for (int k = 0; k < 35; ++k) sacc += fn[k] * p.dtw_W1[n * 35 + k];
      h1[n] = fmaxf(sacc, 0.0f);
    }
    float h2[32];
    for (int n = 0; n < 32; ++n) {
      float sacc = p.dtw_b2[n];
      for (int k = 0; k < 64; ++k) sacc += h1[k] * p.dtw_W2[n * 64 + k];
      h2[n] = fmaxf(sacc, 0.0f);
    }
    float best = -3.4e38f; int bi = 0;
    for (int n = 0; n < 14; ++n) {
      float sacc = p.dtw_b3[n];
      for (int k = 0; k < 32; ++k) sacc += h2[k] * p.dtw_W3[n * 32 + k];
      if (sacc > best) { best = sacc; bi = n; }
    }
    const int wl = bi + 2;
    s_soff[r] = (wl - 1) >> 1;
    s_eoff[r] = wl >> 1;
  }
  __syncthreads();

  // ---------------- Phase 4: forward / backward window GRUs ----------------
  v8f hf[4], hb[4];
  run_gru8(p, p.w_gfWih, p.w_gfWhh, p.gf_bih, p.gf_bhh, s_soff, true, s_h,
           lane, b0, hf);
  run_gru8(p, p.w_gbWih, p.w_gbWhh, p.gb_bih, p.gb_bhh, s_eoff, false, s_h,
           lane, b0, hb);
  __syncthreads();
#pragma unroll
  for (int tt = 0; tt < 4; ++tt) {
    store_dfrag_bf16(s_comb, 128, tt * 16, hf[tt], lane);
    store_dfrag_bf16(s_comb, 128, 64 + tt * 16, hb[tt], lane);
  }
  __syncthreads();

  // ---------------- Phase 5: bi_mlp + mu / sigma heads ---------------------
  {
    v8f H1[4];
#pragma unroll
    for (int nt = 0; nt < 4; ++nt) H1[nt] = splat8(p.bi_b1[nt * 16 + (lane & 15)]);
#pragma unroll
    for (int kt = 0; kt < 4; ++kt) {
      const v16bf A = load_afrag_lds(s_comb, 128, kt, lane);
#pragma unroll
      for (int nt = 0; nt < 4; ++nt)
        H1[nt] = wmma_bf(A, load_bfrag(p.w_biW1, 128, nt, kt, lane), H1[nt]);
    }
#pragma unroll
    for (int nt = 0; nt < 4; ++nt)
#pragma unroll
      for (int j = 0; j < 8; ++j) H1[nt][j] = fmaxf(H1[nt][j], 0.0f);
    __syncthreads();
#pragma unroll
    for (int nt = 0; nt < 4; ++nt) store_dfrag_bf16(s_tmp, 64, nt * 16, H1[nt], lane);
    __syncthreads();

    v8f Z[4];
#pragma unroll
    for (int nt = 0; nt < 4; ++nt) Z[nt] = splat8(p.bi_b2[nt * 16 + (lane & 15)]);
#pragma unroll
    for (int kt = 0; kt < 2; ++kt) {
      const v16bf A = load_afrag_lds(s_tmp, 64, kt, lane);
#pragma unroll
      for (int nt = 0; nt < 4; ++nt)
        Z[nt] = wmma_bf(A, load_bfrag(p.w_biW2, 64, nt, kt, lane), Z[nt]);
    }
    __syncthreads();
#pragma unroll
    for (int nt = 0; nt < 4; ++nt) store_dfrag_bf16(s_h, 64, nt * 16, Z[nt], lane);
    __syncthreads();

    v8f MU[4], LV[4];
#pragma unroll
    for (int nt = 0; nt < 4; ++nt) {
      MU[nt] = splat8(p.mu_b[nt * 16 + (lane & 15)]);
      LV[nt] = splat8(p.lv_b[nt * 16 + (lane & 15)]);
    }
#pragma unroll
    for (int kt = 0; kt < 2; ++kt) {
      const v16bf A = load_afrag_lds(s_h, 64, kt, lane);
#pragma unroll
      for (int nt = 0; nt < 4; ++nt) {
        MU[nt] = wmma_bf(A, load_bfrag(p.w_muW, 64, nt, kt, lane), MU[nt]);
        LV[nt] = wmma_bf(A, load_bfrag(p.w_lvW, 64, nt, kt, lane), LV[nt]);
      }
    }

    const size_t BN = (size_t)p.B * 64;
    const int roff = (lane & 16) ? 8 : 0;
#pragma unroll
    for (int nt = 0; nt < 4; ++nt) {
      const int col = nt * 16 + (lane & 15);
#pragma unroll
      for (int j = 0; j < 8; ++j) {
        const size_t idx = (size_t)(b0 + roff + j) * 64 + col;
        const float mu = MU[nt][j];
        p.out[idx] = mu;
        p.out[BN + idx] = mu;
        p.out[2 * BN + idx] = __expf(0.5f * LV[nt][j]);
      }
    }
  }
}

// ---------------------------------------------------------------------------
// Weight f32 -> bf16 conversion (prep pass into d_ws)
// ---------------------------------------------------------------------------
__global__ void smpe2_cvt_bf16(const float* __restrict__ src,
                               bf16* __restrict__ dst, int n) {
  const int i = blockIdx.x * 256 + threadIdx.x;
  if (i < n) dst[i] = (bf16)src[i];
}

// ws layout (bf16 elements; all 16B aligned)
static constexpr size_t OFF_HIST_WIH = 0;        // 96*160  = 15360
static constexpr size_t OFF_HIST_WHH = 15360;    // 96*32   = 3072
static constexpr size_t OFF_GF_WIH   = 18432;    // 192*160 = 30720
static constexpr size_t OFF_GF_WHH   = 49152;    // 192*64  = 12288
static constexpr size_t OFF_GB_WIH   = 61440;    // 30720
static constexpr size_t OFF_GB_WHH   = 92160;    // 12288
static constexpr size_t OFF_BI_W1    = 104448;   // 64*128  = 8192
static constexpr size_t OFF_BI_W2    = 112640;   // 64*64   = 4096
static constexpr size_t OFF_MU_W     = 116736;   // 4096
static constexpr size_t OFF_LV_W     = 120832;   // 4096  -> total 124928 bf16

extern "C" void kernel_launch(void* const* d_in, const int* in_sizes, int n_in,
                              void* d_out, int out_size, void* d_ws,
                              size_t ws_size, hipStream_t stream) {
  (void)n_in; (void)out_size; (void)ws_size;
  const float* obs      = (const float*)d_in[0];
  const float* act      = (const float*)d_in[1];
  const float* histWih  = (const float*)d_in[2];
  const float* histWhh  = (const float*)d_in[3];
  const float* hist_bih = (const float*)d_in[4];
  const float* hist_bhh = (const float*)d_in[5];
  const float* ln_g     = (const float*)d_in[6];
  const float* ln_b     = (const float*)d_in[7];
  const float* dtw_W1   = (const float*)d_in[8];
  const float* dtw_b1   = (const float*)d_in[9];
  const float* dtw_W2   = (const float*)d_in[10];
  const float* dtw_b2   = (const float*)d_in[11];
  const float* dtw_W3   = (const float*)d_in[12];
  const float* dtw_b3   = (const float*)d_in[13];
  const float* gfWih    = (const float*)d_in[14];
  const float* gfWhh    = (const float*)d_in[15];
  const float* gf_bih   = (const float*)d_in[16];
  const float* gf_bhh   = (const float*)d_in[17];
  const float* gbWih    = (const float*)d_in[18];
  const float* gbWhh    = (const float*)d_in[19];
  const float* gb_bih   = (const float*)d_in[20];
  const float* gb_bhh   = (const float*)d_in[21];
  const float* biW1     = (const float*)d_in[22];
  const float* bi_b1    = (const float*)d_in[23];
  const float* biW2     = (const float*)d_in[24];
  const float* bi_b2    = (const float*)d_in[25];
  const float* muW      = (const float*)d_in[26];
  const float* mu_b     = (const float*)d_in[27];
  const float* lvW      = (const float*)d_in[28];
  const float* lv_b     = (const float*)d_in[29];

  bf16* ws = (bf16*)d_ws;
  auto cvt = [&](const float* s, size_t off, int n) {
    smpe2_cvt_bf16<<<(n + 255) / 256, 256, 0, stream>>>(s, ws + off, n);
  };
  cvt(histWih, OFF_HIST_WIH, 96 * 160);
  cvt(histWhh, OFF_HIST_WHH, 96 * 32);
  cvt(gfWih,   OFF_GF_WIH,   192 * 160);
  cvt(gfWhh,   OFF_GF_WHH,   192 * 64);
  cvt(gbWih,   OFF_GB_WIH,   192 * 160);
  cvt(gbWhh,   OFF_GB_WHH,   192 * 64);
  cvt(biW1,    OFF_BI_W1,    64 * 128);
  cvt(biW2,    OFF_BI_W2,    64 * 64);
  cvt(muW,     OFF_MU_W,     64 * 64);
  cvt(lvW,     OFF_LV_W,     64 * 64);

  SmpeParams p;
  p.obs = obs; p.act = act;
  p.hist_bih = hist_bih; p.hist_bhh = hist_bhh;
  p.ln_g = ln_g; p.ln_b = ln_b;
  p.dtw_W1 = dtw_W1; p.dtw_b1 = dtw_b1;
  p.dtw_W2 = dtw_W2; p.dtw_b2 = dtw_b2;
  p.dtw_W3 = dtw_W3; p.dtw_b3 = dtw_b3;
  p.gf_bih = gf_bih; p.gf_bhh = gf_bhh;
  p.gb_bih = gb_bih; p.gb_bhh = gb_bhh;
  p.bi_b1 = bi_b1; p.bi_b2 = bi_b2;
  p.mu_b = mu_b; p.lv_b = lv_b;
  p.w_histWih = ws + OFF_HIST_WIH; p.w_histWhh = ws + OFF_HIST_WHH;
  p.w_gfWih = ws + OFF_GF_WIH;     p.w_gfWhh = ws + OFF_GF_WHH;
  p.w_gbWih = ws + OFF_GB_WIH;     p.w_gbWhh = ws + OFF_GB_WHH;
  p.w_biW1 = ws + OFF_BI_W1;       p.w_biW2 = ws + OFF_BI_W2;
  p.w_muW = ws + OFF_MU_W;         p.w_lvW = ws + OFF_LV_W;
  p.out = (float*)d_out;
  p.B = in_sizes[0] / (22 * 128);

  smpe2_main<<<p.B / 16, 32, 0, stream>>>(p);
}
